// MambaRegressor_79182017069617
// MI455X (gfx1250) — matrix operand order repaired and checked by reading, not compiled, not run on hardware
//
#include <hip/hip_runtime.h>
#include <hip/hip_bf16.h>
#include <math.h>

typedef _Float16 f16;
typedef __attribute__((ext_vector_type(16))) _Float16 v16h;
typedef __attribute__((ext_vector_type(8)))  _Float16 v8h;
typedef __attribute__((ext_vector_type(8)))  float    v8f;
typedef __attribute__((ext_vector_type(4)))  unsigned int v4u;
typedef __attribute__((ext_vector_type(8)))  int      v8i;
typedef __attribute__((ext_vector_type(4)))  int      v4i;

#define B_      32
#define K_      256
#define DIN_    512
#define PROJ_   64
#define DM_     128
#define NL_     3
#define PL_     8
#define ST_     4
#define NPAT_   63
#define DI_     256
#define DSTATE_ 16
#define DCONV_  4
#define DTRANK_ 8
#define SEQS_   (B_*PROJ_)        /* 2048 */
#define MTOK_   (SEQS_*NPAT_)     /* 129024 */
#define DBLW_   (DTRANK_ + 2*DSTATE_)  /* 40 */

#if defined(__has_builtin)
# if __has_builtin(__builtin_amdgcn_tensor_load_to_lds)
#  define HAVE_TDM 1
# endif
#endif
#ifndef HAVE_TDM
# define HAVE_TDM 0
#endif

// ---------------------------------------------------------------- converts / packing
__global__ void f32_to_f16_kernel(const float* __restrict__ src,
                                  f16* __restrict__ dst, int n) {
  int i = blockIdx.x * 256 + threadIdx.x;
  if (i < n) dst[i] = (f16)src[i];
}

// B[K][N] (f32, row-major) -> Bt[NPAD][K] (f16, transposed, zero-padded rows)
__global__ void pack_bt_kernel(const float* __restrict__ B, f16* __restrict__ Bt,
                               int Kd, int Nd, int NPAD) {
  int idx = blockIdx.x * 256 + threadIdx.x;
  if (idx >= Kd * NPAD) return;
  int k = idx % Kd, n = idx / Kd;
  Bt[(size_t)n * Kd + k] = (f16)((n < Nd) ? B[(size_t)k * Nd + n] : 0.0f);
}

// ---------------------------------------------------------------- TDM helper
#if HAVE_TDM
// Issue a TDM 2D tile load: 64 rows x 32 f16 from A[row_stride=K elems] into LDS
// with 8-elem (4-DWORD) padding after each 32-elem (16-DWORD) row -> LDS stride 40.
__device__ __forceinline__ void tdm_load_tile(const f16* gptr, unsigned lds_off,
                                              int M, int Kel) {
  unsigned long long ga = (unsigned long long)(uintptr_t)gptr;
  v4u g0;
  g0[0] = 1u;                                   // count=1, user mode, no gather
  g0[1] = lds_off;                              // lds_addr
  g0[2] = (unsigned)ga;                         // global_addr[31:0]
  g0[3] = (unsigned)((ga >> 32) & 0x1FFFFFFu) | (2u << 30);  // addr[56:32] | type=2
  v8i g1;
  g1[0] = (int)((1u << 16)   /* data_size = 2B */
              | (1u << 20)   /* pad_enable */
              | (3u << 22)   /* pad_interval: 16 DWORDs */
              | (3u << 25)); /* pad_amount: 4 DWORDs */
  g1[1] = (int)(((unsigned)Kel & 0xFFFFu) << 16);               // tensor_dim0 lo
  g1[2] = (int)(((unsigned)Kel >> 16) | (((unsigned)M & 0xFFFFu) << 16)); // dim0 hi | dim1 lo
  g1[3] = (int)(((unsigned)M >> 16) | (32u << 16));             // dim1 hi | tile_dim0=32
  g1[4] = 64;                                                   // tile_dim1=64
  g1[5] = Kel;                                                  // tensor_dim0_stride lo
  g1[6] = 0;
  g1[7] = 0;
  v4i z4 = {};
#if __clang_major__ >= 23
  v8i z8 = {};
  __builtin_amdgcn_tensor_load_to_lds(g0, g1, z4, z4, z8, 0);
#else
  __builtin_amdgcn_tensor_load_to_lds(g0, g1, z4, z4, 0);
#endif
}
#endif

// ---------------------------------------------------------------- WMMA GEMM
// C[M,N] = A[M,K] * B[K,N] (+bias); A: f16 row-major, Bt: f16 [NPAD][K] transposed.
// block: 256 threads = 8 waves, 4(M) x 2(N); block tile 64x64; wave tile 16x32.
// A tiles stream via TDM (double-buffered, TENSORcnt), B tiles via b128 loads.
template<int N, int NPAD, int K>
__global__ __launch_bounds__(256) void gemm_wmma(
    const f16* __restrict__ A, const f16* __restrict__ Bt,
    const float* __restrict__ bias,
    float* __restrict__ Cout, float* __restrict__ Cacc, int M)
{
  constexpr int TM = 64, TN = 64, TK = 32, LDT = TK + 8;  // 80B rows, 16B aligned
  constexpr int NK = K / TK;
  __shared__ __align__(16) _Float16 As[2][TM][LDT];
  __shared__ __align__(16) _Float16 Bs[2][TN][LDT];
  const int tid  = threadIdx.x;
  const int m0   = blockIdx.x * TM;
  const int n0   = blockIdx.y * TN;
  const int wave = tid >> 5, lane = tid & 31;
  const int wm   = wave >> 1, wn = wave & 1;
  const int lr   = lane & 15, lh = lane >> 4;

  // keep As visibly written so the TDM-filled LDS reads can't be folded away
  if (M == -12345) As[0][0][tid & 31] = (f16)0.0f;

  v8f acc0 = {}, acc1 = {};

#if HAVE_TDM
  if (wave == 0)
    tdm_load_tile(A + (size_t)m0 * K, (unsigned)(uintptr_t)&As[0][0][0], M, K);
#endif

  #pragma unroll
  for (int i = 0; i < NK; ++i) {
    const int k0 = i * TK;
    const int buf = i & 1;
    { // B tile: 64 rows (n) x 32 (k) from transposed weights, one b128 each
      int r = tid >> 2, c = (tid & 3) * 8;
      *reinterpret_cast<v8h*>(&Bs[buf][r][c]) =
          *reinterpret_cast<const v8h*>(Bt + (size_t)(n0 + r) * K + (k0 + c));
    }
#if HAVE_TDM
    if (wave == 0) {
      if (i + 1 < NK) {
        tdm_load_tile(A + (size_t)m0 * K + (k0 + TK),
                      (unsigned)(uintptr_t)&As[(i + 1) & 1][0][0], M, K);
        __builtin_amdgcn_s_wait_tensorcnt(1);   // tile i has landed
      } else {
        __builtin_amdgcn_s_wait_tensorcnt(0);
      }
    }
#else
    { // fallback: cooperative A-tile load
      int r = tid >> 2, c = (tid & 3) * 8;
      *reinterpret_cast<v8h*>(&As[buf][r][c]) =
          *reinterpret_cast<const v8h*>(A + (size_t)(m0 + r) * K + (k0 + c));
      if (k0 + TK < K)
        __builtin_prefetch(A + (size_t)(m0 + r) * K + (k0 + TK + c), 0, 1);
    }
#endif
    __syncthreads();
    // fragments in contiguous k order (k = 16*lh + s), same for A and B (b128 LDS loads)
    v16h af, bf0, bf1;
    {
      const _Float16* ap = &As[buf][wm * 16 + lr][0];
      v8h a0 = *reinterpret_cast<const v8h*>(ap + lh * 16);      // K 0-7 / 16-23
      v8h a1 = *reinterpret_cast<const v8h*>(ap + lh * 16 + 8);  // K 8-15 / 24-31
      const _Float16* bp0 = &Bs[buf][wn * 32 + lr][0];
      const _Float16* bp1 = &Bs[buf][wn * 32 + 16 + lr][0];
      v8h b0 = *reinterpret_cast<const v8h*>(bp0 + lh * 16);     // K 0-15 / 16-31
      v8h b1 = *reinterpret_cast<const v8h*>(bp0 + lh * 16 + 8);
      v8h b2 = *reinterpret_cast<const v8h*>(bp1 + lh * 16);
      v8h b3 = *reinterpret_cast<const v8h*>(bp1 + lh * 16 + 8);
      #pragma unroll
      for (int e = 0; e < 8; ++e) {
        af[e] = a0[e]; af[e + 8] = a1[e];
        bf0[e] = b0[e]; bf0[e + 8] = b1[e];
        bf1[e] = b2[e]; bf1[e + 8] = b3[e];
      }
    }
    acc0 = __builtin_amdgcn_wmma_f32_16x16x32_f16(
        false, af, false, bf0, (short)0, acc0, false, false);
    acc1 = __builtin_amdgcn_wmma_f32_16x16x32_f16(
        false, af, false, bf1, (short)0, acc1, false, false);
    __syncthreads();
  }

  // C/D layout: lane 0-15 -> M=v, lane 16-31 -> M=8+v ; N = lane&15
  #pragma unroll
  for (int h2 = 0; h2 < 2; ++h2) {
    const v8f& acc = h2 ? acc1 : acc0;
    #pragma unroll
    for (int v = 0; v < 8; ++v) {
      int row = m0 + wm * 16 + lh * 8 + v;
      int col = n0 + wn * 32 + h2 * 16 + lr;
      if (col < N) {
        float val = acc[v];
        if (bias) val += bias[col];
        size_t idx = (size_t)row * N + col;
        if (Cacc) Cacc[idx] += val;
        else      Cout[idx]  = val;
      }
    }
  }
}

// ---------------------------------------------------------------- patch embed
__global__ void embed_kernel(const float* __restrict__ H, const float* __restrict__ We,
                             const float* __restrict__ be, float* __restrict__ U) {
  int idx = blockIdx.x * 256 + threadIdx.x;       // exactly MTOK_*DM_ threads
  int dm = idx & (DM_ - 1);
  int m  = idx >> 7;
  int np = m % NPAT_;
  int bp = m / NPAT_;
  int pj = bp & (PROJ_ - 1);
  int b  = bp >> 6;
  float acc = be[dm];
  #pragma unroll
  for (int pl = 0; pl < PL_; ++pl)
    acc += H[((size_t)b * K_ + np * ST_ + pl) * PROJ_ + pj] * We[pl * DM_ + dm];
  U[idx] = acc;
}

// ---------------------------------------------------------------- rmsnorm
__global__ __launch_bounds__(256) void rmsnorm_kernel(
    const float* __restrict__ U, const float* __restrict__ w,
    f16* __restrict__ outh, float* __restrict__ outf) {
  int wave = threadIdx.x >> 5, lane = threadIdx.x & 31;
  int row  = blockIdx.x * 8 + wave;
  const float4 xv = *reinterpret_cast<const float4*>(U + (size_t)row * DM_ + lane * 4);
  float ss = xv.x * xv.x + xv.y * xv.y + xv.z * xv.z + xv.w * xv.w;
  #pragma unroll
  for (int off = 16; off > 0; off >>= 1) ss += __shfl_xor(ss, off, 32);
  float scale = rsqrtf(ss * (1.0f / DM_) + 1e-5f);
  const float xs[4] = {xv.x, xv.y, xv.z, xv.w};
  #pragma unroll
  for (int j = 0; j < 4; ++j) {
    int c = lane * 4 + j;
    float val = xs[j] * scale * w[c];
    if (outh) outh[(size_t)row * DM_ + c] = (f16)val;
    if (outf) outf[(size_t)row * DM_ + c] = val;
  }
}

// ---------------------------------------------------------------- conv + silu
__global__ void conv_silu_kernel(const float* __restrict__ XZ, const float* __restrict__ cw,
                                 const float* __restrict__ cb,
                                 float* __restrict__ XHf, f16* __restrict__ XHh) {
  int idx = blockIdx.x * 256 + threadIdx.x;       // exactly MTOK_*DI_ threads
  int c   = idx & (DI_ - 1);
  int m   = idx >> 8;
  int t   = m % NPAT_;
  int seq = m / NPAT_;
  float acc = cb[c];
  #pragma unroll
  for (int j = 0; j < DCONV_; ++j) {
    int tt = t - (DCONV_ - 1) + j;
    if (tt >= 0)
      acc += XZ[((size_t)seq * NPAT_ + tt) * (2 * DI_) + c] * cw[c * DCONV_ + j];
  }
  float s = acc / (1.0f + __expf(-acc));
  XHf[idx] = s;
  XHh[idx] = (f16)s;
}

// ---------------------------------------------------------------- selective scan
__global__ __launch_bounds__(256) void scan_kernel(
    const float* __restrict__ DBL, const float* __restrict__ XZ,
    const float* __restrict__ XHf, const float* __restrict__ Wdt,
    const float* __restrict__ bdt, const float* __restrict__ A_log,
    const float* __restrict__ Dp, f16* __restrict__ Yh) {
  __shared__ float dbl_s[NPAT_ * DBLW_];
  const int seq = blockIdx.x, di = threadIdx.x;
  for (int i = threadIdx.x; i < NPAT_ * DBLW_; i += 256)
    dbl_s[i] = DBL[(size_t)seq * NPAT_ * DBLW_ + i];
  float wdt_r[DTRANK_];
  #pragma unroll
  for (int r = 0; r < DTRANK_; ++r) wdt_r[r] = Wdt[r * DI_ + di];
  const float bdtv = bdt[di], dp = Dp[di];
  float a_r[DSTATE_], h[DSTATE_];
  #pragma unroll
  for (int s = 0; s < DSTATE_; ++s) {
    a_r[s] = -__expf(A_log[di * DSTATE_ + s]);
    h[s] = 0.0f;
  }
  __syncthreads();
  for (int t = 0; t < NPAT_; ++t) {
    const float* row = dbl_s + t * DBLW_;
    float dtv = bdtv;
    #pragma unroll
    for (int r = 0; r < DTRANK_; ++r) dtv += row[r] * wdt_r[r];
    float delta = (dtv > 20.0f) ? dtv : log1pf(__expf(dtv));
    size_t tok = (size_t)seq * NPAT_ + t;
    float xh  = XHf[tok * DI_ + di];
    float dxu = delta * xh;
    float y = 0.0f;
    #pragma unroll
    for (int s = 0; s < DSTATE_; ++s) {
      float hs = h[s] * __expf(delta * a_r[s]) + dxu * row[DTRANK_ + s];
      h[s] = hs;
      y += hs * row[DTRANK_ + DSTATE_ + s];
    }
    float z   = XZ[tok * (2 * DI_) + DI_ + di];
    float out = (y + xh * dp) * (z / (1.0f + __expf(-z)));
    Yh[tok * DI_ + di] = (f16)out;
  }
}

// ---------------------------------------------------------------- heads
__global__ __launch_bounds__(256) void head1_kernel(
    const float* __restrict__ Uf, const float* __restrict__ hw,
    const float* __restrict__ hb, float* __restrict__ y1) {
  __shared__ float red[8];
  int bp = blockIdx.x;
  float p = 0.0f;
  for (int i = threadIdx.x; i < NPAT_ * DM_; i += 256)
    p += Uf[(size_t)bp * NPAT_ * DM_ + i] * hw[i];
  #pragma unroll
  for (int off = 16; off > 0; off >>= 1) p += __shfl_xor(p, off, 32);
  int wave = threadIdx.x >> 5, lane = threadIdx.x & 31;
  if (lane == 0) red[wave] = p;
  __syncthreads();
  if (threadIdx.x == 0) {
    float s = hb[0];
    #pragma unroll
    for (int wv = 0; wv < 8; ++wv) s += red[wv];
    y1[bp] = s;
  }
}

__global__ void head2_kernel(const float* __restrict__ y1, const float* __restrict__ Wh,
                             const float* __restrict__ bh, float* __restrict__ out) {
  int tid = threadIdx.x;
  if (tid < B_ * 2) {
    int b = tid >> 1, c = tid & 1;
    float s = bh[c];
    #pragma unroll
    for (int pj = 0; pj < PROJ_; ++pj) s += y1[b * PROJ_ + pj] * Wh[pj * 2 + c];
    out[tid] = s;
  }
}

// ---------------------------------------------------------------- launch
extern "C" void kernel_launch(void* const* d_in, const int* in_sizes, int n_in,
                              void* d_out, int out_size, void* d_ws, size_t ws_size,
                              hipStream_t stream) {
  (void)in_sizes; (void)n_in; (void)out_size; (void)ws_size;
  const float* x        = (const float*)d_in[0];
  const float* W_proj   = (const float*)d_in[1];
  const float* b_proj   = (const float*)d_in[2];
  const float* W_embed  = (const float*)d_in[3];
  const float* b_embed  = (const float*)d_in[4];
  const float* norm_w   = (const float*)d_in[5];
  const float* in_proj  = (const float*)d_in[6];
  const float* conv_w   = (const float*)d_in[7];
  const float* conv_b   = (const float*)d_in[8];
  const float* x_proj   = (const float*)d_in[9];
  const float* dt_w     = (const float*)d_in[10];
  const float* dt_b     = (const float*)d_in[11];
  const float* A_log    = (const float*)d_in[12];
  const float* Dp       = (const float*)d_in[13];
  const float* out_proj = (const float*)d_in[14];
  const float* fnorm_w  = (const float*)d_in[15];
  const float* hf_w     = (const float*)d_in[16];
  const float* hf_b     = (const float*)d_in[17];
  const float* W_head   = (const float*)d_in[18];
  const float* b_head   = (const float*)d_in[19];

  char* base = (char*)d_ws;
  size_t off = 0;
  auto carve = [&](size_t bytes) -> void* {
    void* p = base + off;
    off += (bytes + 255) & ~(size_t)255;
    return p;
  };
  const int MX = B_ * K_;                         // 8192 rows for GEMM1
  f16*   Xh   = (f16*)  carve((size_t)MX * DIN_ * 2);
  f16*   WpT  = (f16*)  carve((size_t)64  * DIN_ * 2);            // [64][512]
  f16*   WiT  = (f16*)  carve((size_t)NL_ * 512 * DM_ * 2);       // [512][128]/layer
  f16*   WxT  = (f16*)  carve((size_t)NL_ * 64  * DI_ * 2);       // [64][256]/layer
  f16*   WoT  = (f16*)  carve((size_t)NL_ * DM_ * DI_ * 2);       // [128][256]/layer
  float* H    = (float*)carve((size_t)MX * PROJ_ * 4);
  float* U    = (float*)carve((size_t)MTOK_ * DM_ * 4);
  f16*   Un   = (f16*)  carve((size_t)MTOK_ * DM_ * 2);
  float* XZ   = (float*)carve((size_t)MTOK_ * 2 * DI_ * 4);
  float* XHf  = (float*)carve((size_t)MTOK_ * DI_ * 4);
  f16*   XHh  = (f16*)  carve((size_t)MTOK_ * DI_ * 2);
  float* DBL  = (float*)carve((size_t)MTOK_ * DBLW_ * 4);
  f16*   Yh   = (f16*)  carve((size_t)MTOK_ * DI_ * 2);
  float* Ufin = (float*)carve((size_t)MTOK_ * DM_ * 4);
  float* y1   = (float*)carve((size_t)SEQS_ * 4);

  f32_to_f16_kernel<<<(MX * DIN_ + 255) / 256, 256, 0, stream>>>(x, Xh, MX * DIN_);
  auto pack = [&](const float* s, f16* d, int Kd, int Nd, int NPAD) {
    pack_bt_kernel<<<(Kd * NPAD + 255) / 256, 256, 0, stream>>>(s, d, Kd, Nd, NPAD);
  };
  pack(W_proj, WpT, DIN_, PROJ_, 64);
  for (int l = 0; l < NL_; ++l) {
    pack(in_proj  + (size_t)l * DM_ * 512,   WiT + (size_t)l * 512 * DM_, DM_, 512, 512);
    pack(x_proj   + (size_t)l * DI_ * DBLW_, WxT + (size_t)l * 64 * DI_,  DI_, DBLW_, 64);
    pack(out_proj + (size_t)l * DI_ * DM_,   WoT + (size_t)l * DM_ * DI_, DI_, DM_, DM_);
  }

  // 1) H = x @ W_proj + b_proj        [8192 x 64], K=512
  gemm_wmma<PROJ_, 64, DIN_><<<dim3(MX / 64, 1), 256, 0, stream>>>(
      Xh, WpT, b_proj, H, nullptr, MX);
  // 2) patch gather + embed -> U      [129024 x 128]
  embed_kernel<<<(MTOK_ * DM_) / 256, 256, 0, stream>>>(H, W_embed, b_embed, U);

  for (int l = 0; l < NL_; ++l) {
    rmsnorm_kernel<<<MTOK_ / 8, 256, 0, stream>>>(U, norm_w + l * DM_, Un, nullptr);
    // XZ = Un @ Wi[l]                 [129024 x 512], K=128
    gemm_wmma<2 * DI_, 512, DM_><<<dim3(MTOK_ / 64, 8), 256, 0, stream>>>(
        Un, WiT + (size_t)l * 512 * DM_, nullptr, XZ, nullptr, MTOK_);
    conv_silu_kernel<<<(MTOK_ * DI_) / 256, 256, 0, stream>>>(
        XZ, conv_w + l * DI_ * DCONV_, conv_b + l * DI_, XHf, XHh);
    // DBL = XH @ Wx[l]                [129024 x 40], K=256
    gemm_wmma<DBLW_, 64, DI_><<<dim3(MTOK_ / 64, 1), 256, 0, stream>>>(
        XHh, WxT + (size_t)l * 64 * DI_, nullptr, DBL, nullptr, MTOK_);
    scan_kernel<<<SEQS_, 256, 0, stream>>>(
        DBL, XZ, XHf, dt_w + l * DTRANK_ * DI_, dt_b + l * DI_,
        A_log + l * DI_ * DSTATE_, Dp + l * DI_, Yh);
    // U += Yh @ Wo[l]                 [129024 x 128], K=256 (residual accumulate)
    gemm_wmma<DM_, DM_, DI_><<<dim3(MTOK_ / 64, 2), 256, 0, stream>>>(
        Yh, WoT + (size_t)l * DM_ * DI_, nullptr, nullptr, U, MTOK_);
  }

  rmsnorm_kernel<<<MTOK_ / 8, 256, 0, stream>>>(U, fnorm_w, nullptr, Ufin);
  head1_kernel<<<SEQS_, 256, 0, stream>>>(Ufin, hf_w, hf_b, y1);
  head2_kernel<<<1, 64, 0, stream>>>(y1, W_head, b_head, (float*)d_out);
}
